// ContrastiveLoss_80479097192773
// MI455X (gfx1250) — matrix operand order repaired and checked by reading, not compiled, not run on hardware
//
#include <hip/hip_runtime.h>
#include <hip/hip_bf16.h>
#include <stdint.h>

#define N_ROWS 8192
#define DIM    1024
#define MARGIN 1.0f
#define NT 2              // N-tiles per wave per K-pass (B regs feed 2 M-tiles)

typedef __attribute__((ext_vector_type(16))) __bf16 bf16x16;
typedef __attribute__((ext_vector_type(8)))  __bf16 bf16x8;
typedef __attribute__((ext_vector_type(8)))  float  f32x8;

union ABFrag { bf16x16 v; bf16x8 h[2]; };

static __device__ __forceinline__ uint16_t f32_to_bf16_rne(float x) {
  uint32_t u = __float_as_uint(x);
  uint32_t r = u + 0x7FFFu + ((u >> 16) & 1u);
  return (uint16_t)(r >> 16);
}

// ---------------- Phase 1: L2-normalize rows, emit bf16 + 1/norm ------------
__global__ __launch_bounds__(256)
void norm_bf16_kernel(const float* __restrict__ src,
                      uint16_t* __restrict__ dst,
                      float* __restrict__ invNorm) {
  const int row = blockIdx.x;
  const int tid = threadIdx.x;
  const float* r = src + (size_t)row * DIM;
  float ss = 0.f;
  for (int j = tid; j < DIM; j += 256) { float v = r[j]; ss += v * v; }
  __shared__ float red[256];
  red[tid] = ss; __syncthreads();
  for (int s = 128; s > 0; s >>= 1) {
    if (tid < s) red[tid] += red[tid + s];
    __syncthreads();
  }
  const float inv = 1.0f / fmaxf(sqrtf(red[0]), 1e-12f);
  if (tid == 0) invNorm[row] = inv;
  uint16_t* d = dst + (size_t)row * DIM;
  for (int j = tid; j < DIM; j += 256) d[j] = f32_to_bf16_rne(r[j] * inv);
}

// ---------------- Phase 2: fp32 diagonal positives --------------------------
__global__ __launch_bounds__(256)
void pos_kernel(const float* __restrict__ img, const float* __restrict__ txt,
                const float* __restrict__ invI, const float* __restrict__ invT,
                float* __restrict__ pos) {
  const int row = blockIdx.x;
  const int tid = threadIdx.x;
  const float* a = img + (size_t)row * DIM;
  const float* b = txt + (size_t)row * DIM;
  float d = 0.f;
  for (int j = tid; j < DIM; j += 256) d += a[j] * b[j];
  __shared__ float red[256];
  red[tid] = d; __syncthreads();
  for (int s = 128; s > 0; s >>= 1) {
    if (tid < s) red[tid] += red[tid + s];
    __syncthreads();
  }
  if (tid == 0) pos[row] = red[0] * invI[row] * invT[row];
}

// ---------------- Phase 3: WMMA GEMM fused with online logsumexp ------------
// Block = 256 threads (8 wave32). Block owns 32 output rows (MT=2 M-tiles,
// A staged once in XOR-swizzled LDS, 64KB). Each wave accumulates NT=2
// N-tiles x 2 M-tiles. B fragments use a depth-4 register rotation (b0..b3,
// 128-element K macro-step): each global load lands in a buffer that is dead
// for the next 3 half-steps (~12 WMMAs of latency cover), and the whole
// working set (~160 VGPRs) stays below the 256-VGPR bank boundary.
__global__ __launch_bounds__(256)
void gemm_lse_kernel(const uint16_t* __restrict__ imgN,
                     const uint16_t* __restrict__ txtN,
                     float* __restrict__ lse) {
  __shared__ uint4 ldsBuf4[4096];                 // 64KB: A tile, later stats
  uint16_t* ldsU16 = (uint16_t*)ldsBuf4;

  const int tid  = threadIdx.x;
  const int wave = tid >> 5;
  const int lane = tid & 31;
  const int l15  = lane & 15;   // A row (low 4b) / B column / C column index
  const int hi   = lane >> 4;   // half-wave selector
  const int m0   = blockIdx.x * 32;

  // Stage 32x1024 bf16 A tile into LDS with 16B-granule XOR swizzle:
  // granule(row, g) stored at row*128 + (g ^ (row & 15)). 16-lane DS reads of
  // a fixed g hit 16 distinct granule%16 classes -> bank-conflict-free.
  {
    const uint4* g = (const uint4*)(imgN + (size_t)m0 * DIM);
    for (int i = tid; i < 32 * 128; i += 256) {
      const int r = i >> 7, gg = i & 127;
      ldsBuf4[r * 128 + (gg ^ (r & 15))] = g[i];
    }
  }
  __syncthreads();

  float runMax[2][8], runSum[2][8];
#pragma unroll
  for (int mt = 0; mt < 2; ++mt)
#pragma unroll
    for (int r = 0; r < 8; ++r) { runMax[mt][r] = -3.0e38f; runSum[mt][r] = 0.f; }

  // One half K-step: A frags for both M-tiles from LDS, 2*NT WMMAs against
  // buffer BBUF, then (optionally) refill BBUF for K-step LOADK.
#define HALF_STEP(KK, BBUF, DO_LOAD, LOADK)                                   \
  {                                                                           \
    const int g0 = ((KK) >> 3) + 2 * hi;                                      \
    ABFrag a0, a1;                                                            \
    a0.h[0] = *(const bf16x8*)(ldsU16 + (((l15)      * 128 + ((g0)     ^ l15)) << 3)); \
    a0.h[1] = *(const bf16x8*)(ldsU16 + (((l15)      * 128 + ((g0 + 1) ^ l15)) << 3)); \
    a1.h[0] = *(const bf16x8*)(ldsU16 + (((l15 + 16) * 128 + ((g0)     ^ l15)) << 3)); \
    a1.h[1] = *(const bf16x8*)(ldsU16 + (((l15 + 16) * 128 + ((g0 + 1) ^ l15)) << 3)); \
    _Pragma("unroll")                                                         \
    for (int j = 0; j < NT; ++j) {                                            \
      acc0[j] = __builtin_amdgcn_wmma_f32_16x16x32_bf16(                      \
          false, a0.v, false, BBUF[j], (short)0, acc0[j], false, false);      \
      acc1[j] = __builtin_amdgcn_wmma_f32_16x16x32_bf16(                      \
          false, a1.v, false, BBUF[j], (short)0, acc1[j], false, false);      \
    }                                                                         \
    if (DO_LOAD) {                                                            \
      _Pragma("unroll")                                                       \
      for (int j = 0; j < NT; ++j)                                            \
        BBUF[j] = *(const bf16x16*)(bptr[j] + (LOADK));                       \
    }                                                                         \
  }

  // 512 N-tiles total; 8 waves x NT tiles per iteration.
  for (int it = 0; it < (N_ROWS / 16) / (8 * NT); ++it) {
    const int tileBase = (it * 8 + wave) * NT;

    f32x8 acc0[NT], acc1[NT];
#pragma unroll
    for (int j = 0; j < NT; ++j)
#pragma unroll
      for (int r = 0; r < 8; ++r) { acc0[j][r] = 0.f; acc1[j][r] = 0.f; }

    // Per-lane B pointers: column n = (tileBase+j)*16 + l15, K offset hi*16.
    const uint16_t* bptr[NT];
#pragma unroll
    for (int j = 0; j < NT; ++j) {
      bptr[j] = txtN + (size_t)((tileBase + j) * 16 + l15) * DIM + hi * 16;
      __builtin_prefetch(bptr[j], 0, 0);   // global_prefetch_b8
    }

    // Depth-4 rotation: b0@K, b1@K+32, b2@K+64, b3@K+96.
    bf16x16 b0[NT], b1[NT], b2[NT], b3[NT];
#pragma unroll
    for (int j = 0; j < NT; ++j) b0[j] = *(const bf16x16*)(bptr[j]);
#pragma unroll
    for (int j = 0; j < NT; ++j) b1[j] = *(const bf16x16*)(bptr[j] + 32);
#pragma unroll
    for (int j = 0; j < NT; ++j) b2[j] = *(const bf16x16*)(bptr[j] + 64);
#pragma unroll
    for (int j = 0; j < NT; ++j) b3[j] = *(const bf16x16*)(bptr[j] + 96);

    for (int k = 0; k < DIM - 128; k += 128) {
      HALF_STEP(k,      b0, true, k + 128)
      HALF_STEP(k + 32, b1, true, k + 160)
      HALF_STEP(k + 64, b2, true, k + 192)
      HALF_STEP(k + 96, b3, true, k + 224)
    }
    HALF_STEP(DIM - 128, b0, false, 0)
    HALF_STEP(DIM - 96,  b1, false, 0)
    HALF_STEP(DIM - 64,  b2, false, 0)
    HALF_STEP(DIM - 32,  b3, false, 0)

    // Online logsumexp fold of the 2*NT 16x16 tiles.
#pragma unroll
    for (int j = 0; j < NT; ++j) {
#pragma unroll
      for (int r = 0; r < 8; ++r) {
        {
          const float v = acc0[j][r];
          const float m = runMax[0][r];
          if (v > m) { runSum[0][r] = runSum[0][r] * __expf(m - v) + 1.0f; runMax[0][r] = v; }
          else       { runSum[0][r] += __expf(v - m); }
        }
        {
          const float v = acc1[j][r];
          const float m = runMax[1][r];
          if (v > m) { runSum[1][r] = runSum[1][r] * __expf(m - v) + 1.0f; runMax[1][r] = v; }
          else       { runSum[1][r] += __expf(v - m); }
        }
      }
    }
  }
#undef HALF_STEP

  // Cross-lane merge within each 16-lane half (rows mt*16+{0..7} in lanes
  // 0-15, rows mt*16+{8..15} in lanes 16-31; xor offsets < 16 stay in-half).
#pragma unroll
  for (int mt = 0; mt < 2; ++mt)
#pragma unroll
    for (int r = 0; r < 8; ++r) {
      float m = runMax[mt][r], s = runSum[mt][r];
      for (int off = 1; off < 16; off <<= 1) {
        const float m2 = __shfl_xor(m, off, 32);
        const float s2 = __shfl_xor(s, off, 32);
        const float mn = fmaxf(m, m2);
        s = s * __expf(m - mn) + s2 * __expf(m2 - mn);
        m = mn;
      }
      runMax[mt][r] = m; runSum[mt][r] = s;
    }

  // All A reads done -> safe to reuse the LDS A buffer for the stats.
  __syncthreads();
  float* sMax = (float*)ldsBuf4;        // [8 waves][32 rows]
  float* sSum = sMax + 256;
  if (l15 == 0) {                        // lanes 0 and 16
#pragma unroll
    for (int mt = 0; mt < 2; ++mt)
#pragma unroll
      for (int r = 0; r < 8; ++r) {
        sMax[wave * 32 + mt * 16 + hi * 8 + r] = runMax[mt][r];
        sSum[wave * 32 + mt * 16 + hi * 8 + r] = runSum[mt][r];
      }
  }
  __syncthreads();

  // Merge across the 8 waves, one row per thread.
  if (tid < 32) {
    float m = sMax[tid], s = sSum[tid];
    for (int w = 1; w < 8; ++w) {
      const float m2 = sMax[w * 32 + tid], s2 = sSum[w * 32 + tid];
      const float mn = fmaxf(m, m2);
      s = s * __expf(m - mn) + s2 * __expf(m2 - mn);
      m = mn;
    }
    lse[m0 + tid] = m + __logf(s);
  }
}

// ---------------- Phase 4: scalar mean ---------------------------------------
__global__ __launch_bounds__(256)
void final_reduce_kernel(const float* __restrict__ lse,
                         const float* __restrict__ pos,
                         float* __restrict__ out) {
  __shared__ float red[256];
  const int tid = threadIdx.x;
  float s = 0.f;
  for (int i = tid; i < N_ROWS; i += 256) s += (lse[i] - MARGIN) - pos[i];
  red[tid] = s; __syncthreads();
  for (int st = 128; st > 0; st >>= 1) {
    if (tid < st) red[tid] += red[tid + st];
    __syncthreads();
  }
  if (tid == 0) out[0] = red[0] / (float)N_ROWS;
}

extern "C" void kernel_launch(void* const* d_in, const int* in_sizes, int n_in,
                              void* d_out, int out_size, void* d_ws, size_t ws_size,
                              hipStream_t stream) {
  const float* img = (const float*)d_in[0];
  const float* txt = (const float*)d_in[1];

  char* ws = (char*)d_ws;
  const size_t embBytes = (size_t)N_ROWS * DIM * sizeof(uint16_t);  // 16 MB
  uint16_t* imgN = (uint16_t*)(ws);
  uint16_t* txtN = (uint16_t*)(ws + embBytes);
  float* invI = (float*)(ws + 2 * embBytes);
  float* invT = invI + N_ROWS;
  float* pos  = invT + N_ROWS;
  float* lse  = pos  + N_ROWS;

  norm_bf16_kernel<<<N_ROWS, 256, 0, stream>>>(img, imgN, invI);
  norm_bf16_kernel<<<N_ROWS, 256, 0, stream>>>(txt, txtN, invT);
  pos_kernel<<<N_ROWS, 256, 0, stream>>>(img, txt, invI, invT, pos);
  gemm_lse_kernel<<<N_ROWS / 32, 256, 0, stream>>>(imgN, txtN, lse);
  final_reduce_kernel<<<1, 256, 0, stream>>>(lse, pos, (float*)d_out);
}
